// SpatialEncoder_19774029431379
// MI455X (gfx1250) — matrix operand order repaired and checked by reading, not compiled, not run on hardware
//
#include <hip/hip_runtime.h>
#include <hip/hip_bf16.h>

// ---------------------------------------------------------------------------
// SpatialEncoder correlation / best-offset selection for MI455X (gfx1250).
//
// Sizes (fixed by the reference harness):
//   SB=2, NV=4, N=512, L=256, H=W=128, padding=5  -> Np=11, 121 offsets,
//   grid offsets are exactly 1-pixel steps -> each point needs a 12x12 patch.
//
// Roofline: latent is 128 MB -> fully resident in the 192 MB L2. Dominant
// traffic is 3072 points x 8 slabs x 18 KB ~ 450 MB of patch reads, almost
// all from L2 after one cold HBM pass (~5.5 us @ 23.3 TB/s). FLOPs (~1 GF)
// are trivial, so the design goal is latency hiding on the gathers.
//
// Strategy (one 256-thread / 8-wave workgroup per (sb, warped-view, point)):
//   1. All threads compute the 256-ch reference feature (4 taps each) -> LDS.
//   2. Loop over 8 chunks of 32 channels, DOUBLE-BUFFERED:
//        - interior points: Tensor-Data-Mover tensor_load_to_lds DMAs the
//          12x12x32 slab for chunk i+1 into buf[1-cur] while the waves
//          compute on buf[cur]; s_wait_tensorcnt(1) (in-order TDM) proves
//          the older slab landed. Border points: clamped manual loads.
//        - each wave owns 16 of the 128 (padded) offsets; builds the f16
//          A-fragment (16 offsets x 32 channels, ISA 16-bit A layout) by
//          4-tap bilinear interp from the LDS patch; B = ref vector
//          replicated across all 16 columns; accumulate with
//          v_wmma_f32_16x16x32_f16  -> D column 0 holds num[j].
//        - den[j] = sum(warp^2) accumulated in f32 VALU alongside.
//   3. score[j] = num[j] / max(sqrt(den[j]), 1e-8)  (ref_n is constant over
//      j so it cancels out of the argmax), argmax (first-max, like jnp).
//   4. Re-gather the winning offset's 256-ch feature in full f32 from global
//      (L2-resident) and write the output.
//
// Only the argmax decision sees f16; output features are full f32.
// ---------------------------------------------------------------------------

#define IMG_W   128
#define IMG_H   128
#define LCH     256
#define NPTS    512
#define NVIEWS  4
#define PAD     5
#define NP      11                 // 2*PAD+1
#define NP2     121                // NP*NP
#define PATCH   12                 // NP+1 texels per axis
#define CK      32                 // channels per chunk == WMMA K
#define SLAB    (CK * PATCH * PATCH)

typedef _Float16 v16h __attribute__((ext_vector_type(16)));
typedef float     v8f __attribute__((ext_vector_type(8)));
typedef unsigned int u32x4 __attribute__((ext_vector_type(4)));
typedef int          i32x4 __attribute__((ext_vector_type(4)));
typedef int          i32x8 __attribute__((ext_vector_type(8)));

#if defined(__gfx1250__) && __has_builtin(__builtin_amdgcn_tensor_load_to_lds) && __has_builtin(__builtin_amdgcn_s_wait_tensorcnt)
#define HAVE_TDM 1
#else
#define HAVE_TDM 0
#endif

__device__ __forceinline__ float clipf(float x, float lo, float hi) {
    return fminf(fmaxf(x, lo), hi);
}
__device__ __forceinline__ int clampi(int x, int lo, int hi) {
    return x < lo ? lo : (x > hi ? hi : x);
}

// 4-tap bilinear sample of one channel plane at unclipped pixel coords (x,y)
__device__ __forceinline__ float bilin_global(const float* __restrict__ plane,
                                              float x, float y) {
    x = clipf(x, 0.f, (float)(IMG_W - 1));
    y = clipf(y, 0.f, (float)(IMG_H - 1));
    float x0f = floorf(x), y0f = floorf(y);
    float wx = x - x0f, wy = y - y0f;
    int x0 = (int)x0f, y0 = (int)y0f;
    int x1 = min(x0 + 1, IMG_W - 1);
    int y1 = min(y0 + 1, IMG_H - 1);
    float v00 = plane[y0 * IMG_W + x0];
    float v01 = plane[y0 * IMG_W + x1];
    float v10 = plane[y1 * IMG_W + x0];
    float v11 = plane[y1 * IMG_W + x1];
    float top = v00 * (1.f - wx) + v01 * wx;
    float bot = v10 * (1.f - wx) + v11 * wx;
    return top * (1.f - wy) + bot * wy;
}

#if HAVE_TDM
// Issue one TDM DMA: 12x12x32 f32 tile (x,y,channel) from a 128x128xC tensor
// into LDS at byte offset `ldsoff`, laid out channel-major [c][y][x].
__device__ __forceinline__ void tdm_load_slab(const float* gsrc, unsigned ldsoff) {
    unsigned long long ga = (unsigned long long)(const void*)gsrc;
    unsigned galo = __builtin_amdgcn_readfirstlane((unsigned)ga);
    unsigned gahi = __builtin_amdgcn_readfirstlane((unsigned)(ga >> 32));
    ldsoff = __builtin_amdgcn_readfirstlane(ldsoff);

    // D# group 0: count=1 | lds_addr | global_addr | type=2 ("image")
    u32x4 g0;
    g0[0] = 1u;
    g0[1] = ldsoff;
    g0[2] = galo;
    g0[3] = (gahi & 0x01FFFFFFu) | (2u << 30);
    // D# group 1: data_size=4B; tensor 128x128; tile 12x12x32;
    // dim0_stride = W (y step), dim1_stride = H*W (channel step)
    i32x8 g1;
    g1[0] = 0x00020000;               // data_size=2 (4 bytes)
    g1[1] = (int)(128u << 16);        // tensor_dim0 lo16 = W
    g1[2] = (int)(128u << 16);        // tensor_dim1 lo16 = H
    g1[3] = (int)(12u  << 16);        // tile_dim0 = 12
    g1[4] = (int)(12u | (32u << 16)); // tile_dim1 = 12, tile_dim2 = 32
    g1[5] = 128;                      // tensor_dim0_stride = W
    g1[6] = 0x40000000;               // tensor_dim1_stride lo16 (=16384) << 16
    g1[7] = 0;                        // tensor_dim1_stride hi
    i32x4 g2 = { 256, 1, 0, 0 };      // tensor_dim2 = 256, dim3 = 1
    i32x4 g3 = { 0, 0, 0, 0 };
    i32x8 g4 = { 0, 0, 0, 0, 0, 0, 0, 0 };  // extra operand (clang-23 6-arg form)
    __builtin_amdgcn_tensor_load_to_lds(g0, g1, g2, g3, g4, 0);
}
#endif

__global__ void __launch_bounds__(256)
spatial_encoder_corr_kernel(const float* __restrict__ uv,
                            const float* __restrict__ latent,
                            const float* __restrict__ latent_scaling,
                            float* __restrict__ out) {
    const int tid   = threadIdx.x;           // 0..255
    const int lane  = tid & 31;              // wave32
    const int wv    = tid >> 5;              // 8 waves
    const int bid   = blockIdx.x;            // SB*3*NPTS blocks
    const int n     = bid % NPTS;
    const int v     = (bid / NPTS) % (NVIEWS - 1);
    const int sb    = bid / (NPTS * (NVIEWS - 1));

    __shared__ float patch[2][SLAB];          // 2 x 18 KB double-buffered slabs
    __shared__ float refv[LCH];               // reference feature (f32)
    __shared__ float num_s[128];
    __shared__ float den_s[128];
    __shared__ int   jbest_s;

    const float lsx = latent_scaling[0];
    const float lsy = latent_scaling[1];

    // ---- base pixel coords (unclipped), matching the reference arithmetic --
    // uvn = uv * (latent_scaling / IMAGE_SIZE) - 1 ; pix = (uvn+1)*0.5*(W-1)
    const float uwx = uv[((size_t)(sb * NVIEWS + 1 + v) * NPTS + n) * 2 + 0];
    const float uwy = uv[((size_t)(sb * NVIEWS + 1 + v) * NPTS + n) * 2 + 1];
    const float xb = (uwx * (lsx / 128.f)) * 0.5f * (float)(IMG_W - 1);
    const float yb = (uwy * (lsy / 128.f)) * 0.5f * (float)(IMG_H - 1);

    const float urx = uv[((size_t)(sb * NVIEWS) * NPTS + n) * 2 + 0];
    const float ury = uv[((size_t)(sb * NVIEWS) * NPTS + n) * 2 + 1];
    const float xr = (urx * (lsx / 128.f)) * 0.5f * (float)(IMG_W - 1);
    const float yr = (ury * (lsy / 128.f)) * 0.5f * (float)(IMG_H - 1);

    const float* __restrict__ img_ref  = latent + (size_t)(sb * NVIEWS) * LCH * IMG_H * IMG_W;
    const float* __restrict__ img_warp = latent + (size_t)(sb * NVIEWS + 1 + v) * LCH * IMG_H * IMG_W;

    // ---- reference feature: one channel per thread ------------------------
    {
        const int c = tid;                       // blockDim == LCH == 256
        refv[c] = bilin_global(img_ref + (size_t)c * IMG_H * IMG_W, xr, yr);
    }

    // ---- patch origin + interior test (workgroup-uniform) -----------------
    const int X0 = (int)floorf(clipf(xb, 0.f, (float)(IMG_W - 1)));
    const int Y0 = (int)floorf(clipf(yb, 0.f, (float)(IMG_H - 1)));
    const bool interior = (X0 - PAD >= 0) && (X0 + PAD + 1 <= IMG_W - 1) &&
                          (Y0 - PAD >= 0) && (Y0 + PAD + 1 <= IMG_H - 1);
    const float* __restrict__ gtile0 =
        img_warp + (size_t)(Y0 - PAD) * IMG_W + (X0 - PAD);   // chunk-0 tile start

    // ---- per-lane offset triples (this wave owns offsets wv*16 .. wv*16+15)
    const int m     = lane & 15;
    const int khalf = lane >> 4;
    const int j     = wv * 16 + m;               // 0..127, valid if < 121
    // reference's gp flattening: p = iy*11+ix, x-offset = iy-5, y-offset = ix-5
    const int dxo = j / NP - PAD;
    const int dyo = j % NP - PAD;

    const float xj = clipf(xb + (float)dxo, 0.f, (float)(IMG_W - 1));
    const float yj = clipf(yb + (float)dyo, 0.f, (float)(IMG_H - 1));
    const float xj0f = floorf(xj), yj0f = floorf(yj);
    float wx = xj - xj0f, wy = yj - yj0f;
    const int xj0 = (int)xj0f, yj0 = (int)yj0f;
    const int xj1 = min(xj0 + 1, IMG_W - 1);
    const int yj1 = min(yj0 + 1, IMG_H - 1);
    // absolute index -> patch slot (patch col t holds clamp(X0-PAD+t))
    const int x0s = clampi(xj0 - (X0 - PAD), 0, PATCH - 1);
    const int x1s = clampi(xj1 - (X0 - PAD), 0, PATCH - 1);
    const int y0s = clampi(yj0 - (Y0 - PAD), 0, PATCH - 1);
    const int y1s = clampi(yj1 - (Y0 - PAD), 0, PATCH - 1);

    float w00 = (1.f - wy) * (1.f - wx);
    float w01 = (1.f - wy) * wx;
    float w10 = wy * (1.f - wx);
    float w11 = wy * wx;
    if (j >= NP2) { w00 = w01 = w10 = w11 = 0.f; }   // zero padded A rows

    v8f   acc  = {};      // WMMA f32 accumulator: D[.,col0] = num[j]
    float dsum = 0.f;     // f32 sum of warp^2 for this lane's 16 channels

#if HAVE_TDM
    // Prologue: kick off the DMA for chunk 0 into buffer 0.
    if (interior && tid < 32) {
        tdm_load_slab(gtile0, (unsigned)(size_t)(void*)&patch[0][0]);
    }
#endif

    for (int cc = 0; cc < LCH; cc += CK) {
        const int cur = (cc >> 5) & 1;           // cc/CK parity
        // All waves are done reading buf[1-cur] (iteration i-1's slab) and,
        // on the first pass, refv is fully populated.
        __syncthreads();

#if HAVE_TDM
        if (interior) {
            if (tid < 32) {
                if (cc + CK < LCH) {
                    // Overlap: DMA chunk i+1 into the other buffer, then wait
                    // until at most 1 TDM op is outstanding -> in-order TDM
                    // guarantees the OLDER slab (buf[cur]) has landed.
                    tdm_load_slab(img_warp + (size_t)(cc + CK) * IMG_H * IMG_W +
                                      (size_t)(Y0 - PAD) * IMG_W + (X0 - PAD),
                                  (unsigned)(size_t)(void*)&patch[1 - cur][0]);
                    __builtin_amdgcn_s_wait_tensorcnt(1);
                } else {
                    __builtin_amdgcn_s_wait_tensorcnt(0);
                }
            }
        } else
#endif
        {
            // border (or no-TDM) path: clamped manual loads, same LDS layout
            for (int i = tid; i < SLAB; i += 256) {
                int c   = i / (PATCH * PATCH);
                int rem = i % (PATCH * PATCH);
                int ry  = rem / PATCH;
                int rx  = rem % PATCH;
                int gy  = clampi(Y0 - PAD + ry, 0, IMG_H - 1);
                int gx  = clampi(X0 - PAD + rx, 0, IMG_W - 1);
                patch[cur][i] = img_warp[(size_t)(cc + c) * IMG_H * IMG_W +
                                         (size_t)gy * IMG_W + gx];
            }
        }
        __syncthreads();   // buf[cur] visible to every wave

        const float* __restrict__ pb = &patch[cur][0];

        // Build A (16x32 f16, ISA 16-bit A layout) and B (ref replicated).
        v16h a, b;
#pragma unroll
        for (int p = 0; p < 8; ++p) {
            // A: lane(m,khalf), VGPR p -> K = 2p + (p>=4 ? 8 : 0) + 8*khalf
            const int ka = 2 * p + (p >= 4 ? 8 : 0) + 8 * khalf;
            const float* P0 = pb + (size_t)ka * (PATCH * PATCH);
            const float* P1 = P0 + (PATCH * PATCH);
            float va0 = w00 * P0[y0s * PATCH + x0s] + w01 * P0[y0s * PATCH + x1s] +
                        w10 * P0[y1s * PATCH + x0s] + w11 * P0[y1s * PATCH + x1s];
            float va1 = w00 * P1[y0s * PATCH + x0s] + w01 * P1[y0s * PATCH + x1s] +
                        w10 * P1[y1s * PATCH + x0s] + w11 * P1[y1s * PATCH + x1s];
            a[2 * p]     = (_Float16)va0;
            a[2 * p + 1] = (_Float16)va1;
            dsum += va0 * va0 + va1 * va1;
            // B: same K mapping as A: K = 2p + (p>=4 ? 8 : 0) + 8*khalf (+h); all 16
            // columns identical (ref vector) so only K mapping matters.
            const int kb = 2 * p + (p >= 4 ? 8 : 0) + 8 * khalf;
            b[2 * p]     = (_Float16)refv[cc + kb];
            b[2 * p + 1] = (_Float16)refv[cc + kb + 1];
        }
        acc = __builtin_amdgcn_wmma_f32_16x16x32_f16(
                  false, a, false, b, (short)0, acc, false, false);
    }

    // ---- fold den across lane halves; publish num (column 0) & den --------
    dsum += __shfl_xor(dsum, 16, 32);
    if (lane < 16) den_s[wv * 16 + m] = dsum;
    if (m == 0) {  // lanes 0 and 16 hold D column 0, rows khalf*8 .. +7
#pragma unroll
        for (int r = 0; r < 8; ++r)
            num_s[wv * 16 + khalf * 8 + r] = acc[r];
    }
    __syncthreads();

    // ---- argmax (first-max, matching jnp.argmax) --------------------------
    if (tid == 0) {
        float best = -1e30f;
        int   bj   = 0;
        for (int jj = 0; jj < NP2; ++jj) {
            float wn = fmaxf(sqrtf(den_s[jj]), 1e-8f);
            float s  = num_s[jj] / wn;
            if (s > best) { best = s; bj = jj; }
        }
        jbest_s = bj;
    }
    __syncthreads();
    const int bj = jbest_s;

    // ---- full-f32 re-gather of winning offset + output --------------------
    {
        const int c = tid;                     // one channel per thread
        const int sdx = bj / NP - PAD;
        const int sdy = bj % NP - PAD;
        float val = bilin_global(img_warp + (size_t)c * IMG_H * IMG_W,
                                 xb + (float)sdx, yb + (float)sdy);
        out[((size_t)(sb * NVIEWS + 1 + v) * LCH + c) * NPTS + n] = val;
        if (v == 0)   // view-0 (reference) output row, written once per sb,n
            out[((size_t)(sb * NVIEWS) * LCH + c) * NPTS + n] = refv[c];
    }
}

extern "C" void kernel_launch(void* const* d_in, const int* in_sizes, int n_in,
                              void* d_out, int out_size, void* d_ws, size_t ws_size,
                              hipStream_t stream) {
    (void)in_sizes; (void)n_in; (void)out_size; (void)d_ws; (void)ws_size;
    const float* uv             = (const float*)d_in[0];
    const float* latent         = (const float*)d_in[1];
    const float* latent_scaling = (const float*)d_in[2];
    // d_in[3] is `padding` (==5); the kernel is specialized for 5 (11x11 grid).
    float* out = (float*)d_out;

    const int SB = 2;
    dim3 grid(SB * (NVIEWS - 1) * NPTS);   // one workgroup per (sb, view, point)
    dim3 block(256);                       // 8 wave32 waves
    spatial_encoder_corr_kernel<<<grid, block, 0, stream>>>(uv, latent,
                                                            latent_scaling, out);
}
